// PathNN_67997922231065
// MI455X (gfx1250) — matrix operand order repaired and checked by reading, not compiled, not run on hardware
//
#include <hip/hip_runtime.h>
#include <hip/hip_bf16.h>

#define N_NODES  20000
#define N_PATHS  100000
#define HN       128
#define N_GRAPHS 128
#define N_CLASSES 10

#define LSTM_WAVES 4          // waves per LSTM workgroup
#define MT 2                  // 16-row M-tiles per wave (32 paths / wave)

typedef __attribute__((ext_vector_type(16))) __bf16 v16bf;
typedef __attribute__((ext_vector_type(8)))  float  v8f;

union ABu { v16bf v; unsigned u[8]; uint4 q[2]; };
union CFu { v8f f; float x[8]; uint4 q[2]; };

__device__ __forceinline__ unsigned short f2bf(float f) {
  unsigned u = __float_as_uint(f);
  unsigned r = u + 0x7fffu + ((u >> 16) & 1u);   // round-to-nearest-even
  return (unsigned short)(r >> 16);
}
__device__ __forceinline__ float sigm(float x) { return 1.0f / (1.0f + __expf(-x)); }

// A-fragment (16x32 bf16, ISA 7.12.2): lane&15 = M row; lane>>4 selects K-half.
// Two 16B chunks per lane: K=kb..kb+7 and K=16+kb..16+kb+7 (kb = 8*(lane>>4)).
__device__ __forceinline__ void load_afrag(ABu& fr, const unsigned short* __restrict__ rowptr, int khalf) {
  const uint4* p = (const uint4*)(rowptr + khalf * 8);
  fr.q[0] = p[0];
  fr.q[1] = p[2];          // +16 shorts = +2 uint4
}

// B-fragment from pre-packed weights: 32B contiguous per lane -> 2x b128.
__device__ __forceinline__ void load_bfrag(ABu& fr, const unsigned* __restrict__ chunk, int lane) {
  const uint4* p = (const uint4*)(chunk + lane * 8);
  fr.q[0] = p[0];
  fr.q[1] = p[1];
}

__device__ __forceinline__ v8f wmma_bf16(const ABu& a, const ABu& b, v8f acc) {
  return __builtin_amdgcn_wmma_f32_16x16x32_bf16(false, a.v, false, b.v, (short)0, acc, false, false);
}

// One 16x16 gate tile pair (2 M-tiles sharing every B fragment), K=128 (+128 recurrent).
__device__ __forceinline__ void gate_pair(v8f& a0, v8f& a1, int gi, int nt, int t,
                                          const unsigned* __restrict__ wlds,
                                          const ABu (&ax)[MT][4], const ABu (&ah)[MT][4],
                                          const float* __restrict__ bsum, int lane, int row) {
  const v8f vzero = {0.f,0.f,0.f,0.f,0.f,0.f,0.f,0.f};
  a0 = vzero; a1 = vzero;
  const int ntile = gi * 8 + nt;
#pragma unroll
  for (int kt = 0; kt < 4; ++kt) {
    ABu bw; load_bfrag(bw, wlds + (size_t)(ntile * 4 + kt) * 256, lane);
    a0 = wmma_bf16(ax[0][kt], bw, a0);
    a1 = wmma_bf16(ax[1][kt], bw, a1);
  }
  if (t > 0) {
#pragma unroll
    for (int kt = 0; kt < 4; ++kt) {
      ABu bw; load_bfrag(bw, wlds + 32768 + (size_t)(ntile * 4 + kt) * 256, lane);
      a0 = wmma_bf16(ah[0][kt], bw, a0);
      a1 = wmma_bf16(ah[1][kt], bw, a1);
    }
  }
  const float bi = bsum[gi * HN + nt * 16 + row];
#pragma unroll
  for (int e = 0; e < 8; ++e) { a0[e] += bi; a1[e] += bi; }
}

// ---------------------------------------------------------------------------
// One LSTM timestep over all paths. 4 waves / workgroup; each wave owns MT=2
// 16-path M-tiles. Weights staged into LDS via gfx1250 async loads (128 KB at
// t=0, 256 KB at t>0); every B fragment feeds 2 WMMAs. LSTM state lives in
// global (L2-resident): c in wave-tile-packed fp32, h in row-major bf16 that
// directly feeds next step's A-fragment loads. Gates computed progressively
// (i -> g -> f -> o) to keep at most ~2 gate tiles live.
// ---------------------------------------------------------------------------
__global__ __launch_bounds__(32 * LSTM_WAVES, 1) void lstm_step_kernel(
    const unsigned short* __restrict__ hbf,    // node features bf16 [N_NODES,128]
    const int* __restrict__ paths, int T, int t,
    const unsigned* __restrict__ wpak,         // wih packed | whh packed
    const float* __restrict__ bsum,            // b_ih + b_hh [512]
    const unsigned short* __restrict__ hprev,  // [N_PATHS,128] bf16 (read when t>0)
    float* __restrict__ cbuf,                  // [N_PATHS*128] fp32, tile-packed
    unsigned short* __restrict__ hnext,        // [N_PATHS,128] bf16 (written when t<T-1)
    float* __restrict__ agg)                   // [N_NODES,128] (atomicAdd when t==T-1)
{
  extern __shared__ unsigned char smem[];
  unsigned* wlds = (unsigned*)smem;
  const int tid = threadIdx.x, lane = tid & 31, wv = tid >> 5;

  // ---- async-stage packed weights into LDS ----
  {
    const int nq = (t > 0) ? 16384 : 8192;               // b128 chunks
    const unsigned smem_base = (unsigned)(size_t)smem;   // LDS offset = low 32 bits
    const unsigned long long gbase = (unsigned long long)(size_t)wpak;
    for (int i = tid; i < nq; i += 32 * LSTM_WAVES) {
      unsigned lds = smem_base + (unsigned)(i * 16);
      unsigned long long g = gbase + (unsigned long long)(i * 16);
      asm volatile("global_load_async_to_lds_b128 %0, %1, off"
                   :: "v"(lds), "v"(g) : "memory");
    }
    asm volatile("s_wait_asynccnt 0" ::: "memory");
  }
  __syncthreads();

  const int gwave = blockIdx.x * LSTM_WAVES + wv;        // 32 paths per wave
  if (gwave * (16 * MT) >= N_PATHS) return;              // after the only barrier
  const int row = lane & 15, khalf = lane >> 4;
  const int pbase0 = gwave * (16 * MT);

  // ---- A fragments: gathered x_t and (t>0) dense h_{t-1} rows ----
  ABu ax[MT][4], ah[MT][4];
#pragma unroll
  for (int mt = 0; mt < MT; ++mt) {
    const int node = paths[(size_t)(pbase0 + mt * 16 + row) * T + t];
    const unsigned short* arow = hbf + (size_t)node * HN;
#pragma unroll
    for (int kt = 0; kt < 4; ++kt) load_afrag(ax[mt][kt], arow + kt * 32, khalf);
    if (t > 0) {
      const unsigned short* hrow = hprev + (size_t)(pbase0 + mt * 16 + row) * HN;
#pragma unroll
      for (int kt = 0; kt < 4; ++kt) load_afrag(ah[mt][kt], hrow + kt * 32, khalf);
    }
  }

#pragma unroll
  for (int nt = 0; nt < 8; ++nt) {
    v8f a0, a1;
    v8f si[MT], ig[MT], cn[MT];

    gate_pair(a0, a1, 0, nt, t, wlds, ax, ah, bsum, lane, row);   // i gate
#pragma unroll
    for (int e = 0; e < 8; ++e) { si[0][e] = sigm(a0[e]); si[1][e] = sigm(a1[e]); }

    gate_pair(a0, a1, 2, nt, t, wlds, ax, ah, bsum, lane, row);   // g gate
#pragma unroll
    for (int e = 0; e < 8; ++e) { ig[0][e] = si[0][e] * tanhf(a0[e]); ig[1][e] = si[1][e] * tanhf(a1[e]); }

    gate_pair(a0, a1, 1, nt, t, wlds, ax, ah, bsum, lane, row);   // f gate
#pragma unroll
    for (int mt = 0; mt < MT; ++mt) {
      CFu cp; cp.f = (v8f){0.f,0.f,0.f,0.f,0.f,0.f,0.f,0.f};
      float* cptr = cbuf + ((size_t)((gwave * MT + mt) * 8 + nt) * 32 + lane) * 8;
      if (t > 0) { cp.q[0] = ((const uint4*)cptr)[0]; cp.q[1] = ((const uint4*)cptr)[1]; }
      const v8f& af = (mt == 0) ? a0 : a1;
#pragma unroll
      for (int e = 0; e < 8; ++e) cn[mt][e] = sigm(af[e]) * cp.x[e] + ig[mt][e];
      if (t + 1 < T) {      // persist c for the next step
        CFu co; co.f = cn[mt];
        ((uint4*)cptr)[0] = co.q[0]; ((uint4*)cptr)[1] = co.q[1];
      }
    }

    gate_pair(a0, a1, 3, nt, t, wlds, ax, ah, bsum, lane, row);   // o gate
#pragma unroll
    for (int mt = 0; mt < MT; ++mt) {
      const v8f& ao = (mt == 0) ? a0 : a1;
#pragma unroll
      for (int e = 0; e < 8; ++e) {
        float hv = sigm(ao[e]) * tanhf(cn[mt][e]);
        const int m = e + khalf * 8;        // C/D layout: M = vgpr (+8 for lanes>=16)
        if (t + 1 < T) {
          hnext[(size_t)(pbase0 + mt * 16 + m) * HN + nt * 16 + row] = f2bf(hv);
        } else {
          const int dn = paths[(size_t)(pbase0 + mt * 16 + m) * T + (T - 1)];
          atomicAdd(&agg[(size_t)dn * HN + nt * 16 + row], hv);
        }
      }
    }
  }
}

// ---------------------------------------------------------------------------
// Weight pre-pack: W [nout,128] fp32 row-major -> B-fragment layout (bf16):
// out[((nt*4+kt)*32 + lane)*8 + v] = pack(W[nt*16+(lane&15)][k0], W[nt*16+(lane&15)][k0+1]),
// k0 = kt*32 + (v>>2)*16 + 8*(lane>>4) + 2*(v&3)
// ---------------------------------------------------------------------------
__global__ void pack_w_kernel(const float* __restrict__ W, unsigned* __restrict__ out, int nout) {
  int idx = blockIdx.x * blockDim.x + threadIdx.x;
  int total = nout * 64;
  if (idx >= total) return;
  int v    = idx & 7;
  int lane = (idx >> 3) & 31;
  int kt   = (idx >> 8) & 3;
  int nt   = idx >> 10;
  int k  = kt * 32 + (v >> 2) * 16 + 8 * (lane >> 4) + 2 * (v & 3);
  int n0 = nt * 16 + (lane & 15);
  unsigned lo = f2bf(W[(size_t)n0 * HN + k]);
  unsigned hi = f2bf(W[(size_t)n0 * HN + k + 1]);
  out[idx] = lo | (hi << 16);
}

__global__ void f2bf_kernel(const float* __restrict__ in, unsigned short* __restrict__ out, int n) {
  int i = blockIdx.x * blockDim.x + threadIdx.x;
  if (i < n) out[i] = f2bf(in[i]);
}

__global__ void addvec_kernel(const float* __restrict__ a, const float* __restrict__ b,
                              float* __restrict__ o, int n) {
  int i = blockIdx.x * blockDim.x + threadIdx.x;
  if (i < n) o[i] = a[i] + b[i];
}

// ---------------------------------------------------------------------------
// Y[M,128] = A_bf16[M,128] @ W^T + bias ; fused BN batch-stat accumulation.
// ---------------------------------------------------------------------------
__global__ __launch_bounds__(256) void gemm_bias_stats_kernel(
    const unsigned short* __restrict__ A, const unsigned* __restrict__ Wp,
    const float* __restrict__ bias, float* __restrict__ Y,
    float* __restrict__ sum, float* __restrict__ sumsq, int M)
{
  const int lane = threadIdx.x & 31;
  const int tile = blockIdx.x * 8 + (threadIdx.x >> 5);
  if (tile >= (M >> 4)) return;
  const int row = lane & 15, khalf = lane >> 4;

  const unsigned short* arow = A + (size_t)(tile * 16 + row) * HN;
  ABu ax[4];
#pragma unroll
  for (int kt = 0; kt < 4; ++kt) load_afrag(ax[kt], arow + kt * 32, khalf);

#pragma unroll
  for (int nt = 0; nt < 8; ++nt) {
    v8f acc = {0.f,0.f,0.f,0.f,0.f,0.f,0.f,0.f};
#pragma unroll
    for (int kt = 0; kt < 4; ++kt) {
      ABu bw; load_bfrag(bw, Wp + (size_t)(nt * 4 + kt) * 256, lane);
      acc = wmma_bf16(ax[kt], bw, acc);
    }
    const int col = nt * 16 + row;           // C/D layout: lane&15 = N column
    const float b = bias[col];
    float s = 0.f, sq = 0.f;
#pragma unroll
    for (int e = 0; e < 8; ++e) {            // VGPR e: M = e (+8 for lane>=16)
      float yv = acc[e] + b;
      Y[(size_t)(tile * 16 + e + khalf * 8) * HN + col] = yv;
      s += yv; sq += yv * yv;
    }
    s  += __shfl_xor(s, 16);                 // combine the two lane halves (wave32)
    sq += __shfl_xor(sq, 16);
    if (khalf == 0) { atomicAdd(&sum[col], s); atomicAdd(&sumsq[col], sq); }
  }
}

__global__ void bn_finalize_kernel(const float* __restrict__ sum, const float* __restrict__ sumsq,
                                   const float* __restrict__ g, const float* __restrict__ be,
                                   float* __restrict__ scale, float* __restrict__ shift, float invM) {
  int j = threadIdx.x;
  float mean = sum[j] * invM;
  float var  = sumsq[j] * invM - mean * mean;
  float sc   = g[j] * rsqrtf(var + 1e-5f);
  scale[j] = sc;
  shift[j] = be[j] - mean * sc;
}

__global__ void bn_apply_kernel(const float* __restrict__ Y, const float* __restrict__ scale,
                                const float* __restrict__ shift, float* __restrict__ hf,
                                unsigned short* __restrict__ hb, int rows) {
  int col = threadIdx.x;
  float sc = scale[col], sh = shift[col];
  int r0 = blockIdx.x * 64, r1 = min(r0 + 64, rows);
  for (int r = r0; r < r1; ++r) {
    float v = Y[(size_t)r * HN + col] * sc + sh;
    v = fmaxf(v, 0.f);
    hf[(size_t)r * HN + col] = v;
    hb[(size_t)r * HN + col] = f2bf(v);
  }
}

__global__ void residual_stats_kernel(const float* __restrict__ agg, const float* __restrict__ h,
                                      float* __restrict__ Y, float* __restrict__ sum,
                                      float* __restrict__ sumsq, int rows) {
  int col = threadIdx.x;
  int r0 = blockIdx.x * 64, r1 = min(r0 + 64, rows);
  float s = 0.f, sq = 0.f;
  for (int r = r0; r < r1; ++r) {
    float v = agg[(size_t)r * HN + col] + h[(size_t)r * HN + col];
    Y[(size_t)r * HN + col] = v;
    s += v; sq += v * v;
  }
  atomicAdd(&sum[col], s); atomicAdd(&sumsq[col], sq);
}

__global__ void pool_kernel(const float* __restrict__ h, const int* __restrict__ batch,
                            float* __restrict__ pooled, int rows) {
  int col = threadIdx.x;
  int r0 = blockIdx.x * 64, r1 = min(r0 + 64, rows);
  float acc = 0.f; int cur = -1;
  for (int r = r0; r < r1; ++r) {
    int b = batch[r];                       // sorted -> long same-graph runs
    if (b != cur) { if (cur >= 0) atomicAdd(&pooled[(size_t)cur * HN + col], acc); cur = b; acc = 0.f; }
    acc += h[(size_t)r * HN + col];
  }
  if (cur >= 0) atomicAdd(&pooled[(size_t)cur * HN + col], acc);
}

__global__ void head_kernel(const float* __restrict__ pooled, const float* __restrict__ w1,
                            const float* __restrict__ b1, const float* __restrict__ w2,
                            const float* __restrict__ b2, float* __restrict__ out) {
  __shared__ float t1[HN];
  const int g = blockIdx.x, j = threadIdx.x;
  float d = b1[j];
  for (int k = 0; k < HN; ++k) d += pooled[(size_t)g * HN + k] * w1[(size_t)j * HN + k];
  t1[j] = fmaxf(d, 0.f);
  __syncthreads();
  if (j < N_CLASSES) {
    float o = b2[j];
    for (int k = 0; k < HN; ++k) o += t1[k] * w2[(size_t)j * HN + k];
    out[(size_t)g * N_CLASSES + j] = o;
  }
}

// ---------------------------------------------------------------------------
extern "C" void kernel_launch(void* const* d_in, const int* in_sizes, int n_in,
                              void* d_out, int out_size, void* d_ws, size_t ws_size,
                              hipStream_t stream) {
  (void)in_sizes; (void)n_in; (void)out_size; (void)ws_size;
  const float* x      = (const float*)d_in[0];
  const int*   path2  = (const int*)d_in[1];
  const int*   path3  = (const int*)d_in[2];
  const int*   batch  = (const int*)d_in[3];
  const float* fe_w1  = (const float*)d_in[4];
  const float* fe_b1  = (const float*)d_in[5];
  const float* fe_g1  = (const float*)d_in[6];
  const float* fe_be1 = (const float*)d_in[7];
  const float* fe_w2  = (const float*)d_in[8];
  const float* fe_b2  = (const float*)d_in[9];
  const float* fe_g2  = (const float*)d_in[10];
  const float* fe_be2 = (const float*)d_in[11];
  const float* w_ih   = (const float*)d_in[12];
  const float* w_hh   = (const float*)d_in[13];
  const float* b_ih   = (const float*)d_in[14];
  const float* b_hh   = (const float*)d_in[15];
  const float* bn1_g  = (const float*)d_in[16];
  const float* bn1_b  = (const float*)d_in[17];
  const float* bn2_g  = (const float*)d_in[18];
  const float* bn2_b  = (const float*)d_in[19];
  const float* lin1_w = (const float*)d_in[20];
  const float* lin1_b = (const float*)d_in[21];
  const float* lin2_w = (const float*)d_in[22];
  const float* lin2_b = (const float*)d_in[23];
  float* out = (float*)d_out;

  // --- workspace carve (~113 MB) ---
  char* base = (char*)d_ws;
  size_t off = 0;
  auto take = [&](size_t bytes) -> void* {
    void* p = base + off;
    off += (bytes + 255) & ~(size_t)255;
    return p;
  };
  float*          Y     = (float*)take((size_t)N_NODES * HN * 4);
  float*          Hf    = (float*)take((size_t)N_NODES * HN * 4);
  unsigned short* Hb    = (unsigned short*)take((size_t)N_NODES * HN * 2); // also x_bf16 initially
  float*          Agg   = (float*)take((size_t)N_NODES * HN * 4);
  unsigned short* Hpath = (unsigned short*)take((size_t)N_PATHS * HN * 2); // LSTM h state
  float*          Cbuf  = (float*)take((size_t)N_PATHS * HN * 4);          // LSTM c state
  unsigned*       wpak  = (unsigned*)take((size_t)2 * 512 * 64 * 4);       // wih | whh contiguous
  unsigned*       w1p   = (unsigned*)take((size_t)128 * 64 * 4);
  unsigned*       w2p   = (unsigned*)take((size_t)128 * 64 * 4);
  float*          bsum  = (float*)take(512 * 4);
  float*          stats = (float*)take(256 * 4);
  float*          scale = (float*)take(128 * 4);
  float*          shift = (float*)take(128 * 4);
  float*          pooled = (float*)take((size_t)N_GRAPHS * HN * 4);
  float* sum = stats;
  float* sumsq = stats + 128;

  // 0) weight pre-pack + bias sum + x -> bf16
  pack_w_kernel<<<(512 * 64 + 255) / 256, 256, 0, stream>>>(w_ih, wpak, 512);
  pack_w_kernel<<<(512 * 64 + 255) / 256, 256, 0, stream>>>(w_hh, wpak + 32768, 512);
  pack_w_kernel<<<(128 * 64 + 255) / 256, 256, 0, stream>>>(fe_w1, w1p, 128);
  pack_w_kernel<<<(128 * 64 + 255) / 256, 256, 0, stream>>>(fe_w2, w2p, 128);
  addvec_kernel<<<2, 256, 0, stream>>>(b_ih, b_hh, bsum, 512);
  f2bf_kernel<<<(N_NODES * HN + 255) / 256, 256, 0, stream>>>(x, Hb, N_NODES * HN);

  const int gemm_grid = (N_NODES / 16 + 7) / 8;
  const int rb_grid   = (N_NODES + 63) / 64;
  const float invM    = 1.0f / (float)N_NODES;

  const int lstm_waves = N_PATHS / (16 * MT);                        // 3125
  const int lstm_grid  = (lstm_waves + LSTM_WAVES - 1) / LSTM_WAVES; // 782
  const size_t lstm_lds = 262144;                                    // 256 KB weights

  // 1) encoder layer 1: Linear -> BN -> ReLU
  hipMemsetAsync(stats, 0, 256 * 4, stream);
  gemm_bias_stats_kernel<<<gemm_grid, 256, 0, stream>>>(Hb, w1p, fe_b1, Y, sum, sumsq, N_NODES);
  bn_finalize_kernel<<<1, 128, 0, stream>>>(sum, sumsq, fe_g1, fe_be1, scale, shift, invM);
  bn_apply_kernel<<<rb_grid, 128, 0, stream>>>(Y, scale, shift, Hf, Hb, N_NODES);

  // 2) encoder layer 2
  hipMemsetAsync(stats, 0, 256 * 4, stream);
  gemm_bias_stats_kernel<<<gemm_grid, 256, 0, stream>>>(Hb, w2p, fe_b2, Y, sum, sumsq, N_NODES);
  bn_finalize_kernel<<<1, 128, 0, stream>>>(sum, sumsq, fe_g2, fe_be2, scale, shift, invM);
  bn_apply_kernel<<<rb_grid, 128, 0, stream>>>(Y, scale, shift, Hf, Hb, N_NODES);

  // 3) PathConv layer 1 (T=2): one kernel per timestep
  hipMemsetAsync(Agg, 0, (size_t)N_NODES * HN * 4, stream);
  lstm_step_kernel<<<lstm_grid, 32 * LSTM_WAVES, lstm_lds, stream>>>(
      Hb, path2, 2, 0, wpak, bsum, Hpath, Cbuf, Hpath, Agg);
  lstm_step_kernel<<<lstm_grid, 32 * LSTM_WAVES, lstm_lds, stream>>>(
      Hb, path2, 2, 1, wpak, bsum, Hpath, Cbuf, Hpath, Agg);
  hipMemsetAsync(stats, 0, 256 * 4, stream);
  residual_stats_kernel<<<rb_grid, 128, 0, stream>>>(Agg, Hf, Y, sum, sumsq, N_NODES);
  bn_finalize_kernel<<<1, 128, 0, stream>>>(sum, sumsq, bn1_g, bn1_b, scale, shift, invM);
  bn_apply_kernel<<<rb_grid, 128, 0, stream>>>(Y, scale, shift, Hf, Hb, N_NODES);

  // 4) PathConv layer 2 (T=3)
  hipMemsetAsync(Agg, 0, (size_t)N_NODES * HN * 4, stream);
  lstm_step_kernel<<<lstm_grid, 32 * LSTM_WAVES, lstm_lds, stream>>>(
      Hb, path3, 3, 0, wpak, bsum, Hpath, Cbuf, Hpath, Agg);
  lstm_step_kernel<<<lstm_grid, 32 * LSTM_WAVES, lstm_lds, stream>>>(
      Hb, path3, 3, 1, wpak, bsum, Hpath, Cbuf, Hpath, Agg);
  lstm_step_kernel<<<lstm_grid, 32 * LSTM_WAVES, lstm_lds, stream>>>(
      Hb, path3, 3, 2, wpak, bsum, Hpath, Cbuf, Hpath, Agg);
  hipMemsetAsync(stats, 0, 256 * 4, stream);
  residual_stats_kernel<<<rb_grid, 128, 0, stream>>>(Agg, Hf, Y, sum, sumsq, N_NODES);
  bn_finalize_kernel<<<1, 128, 0, stream>>>(sum, sumsq, bn2_g, bn2_b, scale, shift, invM);
  bn_apply_kernel<<<rb_grid, 128, 0, stream>>>(Y, scale, shift, Hf, Hb, N_NODES);

  // 5) global add pool + MLP head
  hipMemsetAsync(pooled, 0, (size_t)N_GRAPHS * HN * 4, stream);
  pool_kernel<<<rb_grid, 128, 0, stream>>>(Hf, batch, pooled, N_NODES);
  head_kernel<<<N_GRAPHS, 128, 0, stream>>>(pooled, lin1_w, lin1_b, lin2_w, lin2_b, out);
}
